// MyMultilayerRNN_75634374082640
// MI455X (gfx1250) — matrix operand order repaired and checked by reading, not compiled, not run on hardware
//
#include <hip/hip_runtime.h>
#include <hip/hip_bf16.h>

// ---------------------------------------------------------------------------
// 2-layer LSTM (T=512, B=64, I=256, H=512) for gfx1250 (MI455X).
//  - bf16 WMMA (v_wmma_f32_16x16x32_bf16) for all gate GEMMs, f32 accumulate.
//  - Weights converted once to bf16, pre-swizzled into WMMA B-fragment order
//    WITH gate-interleaved columns (i,f,g,o for the same hidden j land in the
//    same 16-wide tile), so the LSTM cell update fuses into the GEMM epilogue
//    via wave32 __shfl (ds_bpermute) -- no gates round-trip, no pointwise
//    kernel, 1 launch per layer per timestep.
//  - Cell nonlinearities use gfx1250 hardware v_tanh_f32 (TRANS32 op) via
//    inline asm; sigmoid(x) = 0.5*tanh(0.5x)+0.5.
//  - All launches on `stream`, no sync/alloc: graph-capture safe.
// ---------------------------------------------------------------------------

#define TT 512
#define BB 64
#define II 256
#define HH 512
#define G4 2048           // 4*H
#define BH (BB * HH)      // 32768

typedef __attribute__((ext_vector_type(16))) __bf16 v16bf;
typedef __attribute__((ext_vector_type(8)))  __bf16 v8bf;
typedef __attribute__((ext_vector_type(8)))  float  v8f;

// ---- workspace layout (bytes) ----
#define OFF_WIH0  0u                 // 2048*256 bf16  = 1,048,576 B
#define OFF_WHH0  1048576u           // 2048*512 bf16  = 2,097,152 B
#define OFF_WIH1  3145728u           // 2048*512 bf16
#define OFF_WHH1  5242880u           // 2048*512 bf16
#define OFF_XB    7340032u           // 512*64*256 bf16 = 16,777,216 B
#define OFF_H0    24117248u          // 64*512 f32
#define OFF_C0    24248320u
#define OFF_H1    24379392u
#define OFF_C1    24510464u
#define OFF_H0B   24641536u          // 64*512 bf16
#define OFF_H1B   24707072u
// total ~ 24.8 MB

__device__ __forceinline__ unsigned short f32_to_bf16_rne(float f) {
  unsigned int u = __float_as_uint(f);
  unsigned int r = u + 0x7FFFu + ((u >> 16) & 1u);
  return (unsigned short)(r >> 16);
}

// gfx1250 hardware tanh (TRANS32). ISA 7.4: TRANS result needs 1 independent
// op before use -> glue an s_delay_alu TRANS32_DEP_1 behind it.
__device__ __forceinline__ float htanh(float x) {
  float y;
  asm volatile("v_tanh_f32 %0, %1\n\t"
               "s_delay_alu instid0(TRANS32_DEP_1)"
               : "=v"(y) : "v"(x));
  return y;
}

__device__ __forceinline__ float hsigmoid(float x) {
  return __builtin_fmaf(0.5f, htanh(0.5f * x), 0.5f);
}

// ---------------------------------------------------------------------------
// Weight conversion + swizzle into WMMA B-fragment order, gate-interleaved.
// Source W: [N=2048][K] row-major f32, gate blocks of 512 (i,f,g,o).
// Swizzled: blocks of (kt,nt) 32x16; block index = nt*(K/32)+kt; within
// block: lane (0..31) * 16 contiguous bf16 values.
//   tile column c = lane&15 : gate g = c>>2, hidden j = nt*4 + (c&3)
//   n_orig = g*512 + j
//   lane<16 : k = kt*32 + v ; lane>=16: k = kt*32 + 16 + v   (v=0..15)
// ---------------------------------------------------------------------------
__global__ void convert_weight_swz(const float* __restrict__ W,
                                   unsigned short* __restrict__ out, int K) {
  int tid = blockIdx.x * blockDim.x + threadIdx.x;
  int total = G4 * K;
  if (tid >= total) return;
  int KT = K >> 5;
  int blk  = tid >> 9;        // /512
  int rem  = tid & 511;
  int lane = rem >> 4;
  int v    = rem & 15;
  int kt = blk % KT;
  int nt = blk / KT;
  int c  = lane & 15;
  int n  = (c >> 2) * HH + nt * 4 + (c & 3);   // gate-interleaved column
  int k  = kt * 32 + ((lane < 16) ? v : (16 + v));
  out[tid] = f32_to_bf16_rne(W[(size_t)n * K + k]);
}

// x f32 [T*B][I] -> bf16 same layout
__global__ void convert_x_bf16(const float* __restrict__ x,
                               unsigned short* __restrict__ out, int total) {
  int tid = blockIdx.x * blockDim.x + threadIdx.x;
  if (tid < total) out[tid] = f32_to_bf16_rne(x[tid]);
}

// zero h/c state (f32) and bf16 h copies
__global__ void init_state(float* h0, float* c0, float* h1, float* c1,
                           unsigned short* h0b, unsigned short* h1b) {
  int idx = blockIdx.x * blockDim.x + threadIdx.x;   // 0..BH-1
  h0[idx] = 0.f; c0[idx] = 0.f; h1[idx] = 0.f; c1[idx] = 0.f;
  h0b[idx] = 0;  h1b[idx] = 0;
}

// ---------------------------------------------------------------------------
// Fused LSTM step (one layer): gates = A1[64][K1]@B1 + A2[64][K2]@B2 (+bias),
// then cell update in the epilogue. A*: bf16 row-major. B*: bf16 pre-swizzled
// gate-interleaved. Grid: 16 blocks x 256 threads (8 waves); wave owns
// n-tile nt = bx*8+wave = 4 hidden indices x 4 gates, across 4 M-tiles.
// Epilogue: per (mt,r), lanes shuffle-gather the 4 gates for (row, j);
// lanes with (lane&15)<4 update c/h, write bf16 h, and (if out) the
// length-masked output[t].
// ---------------------------------------------------------------------------
__global__ __launch_bounds__(256)
void lstm_step_layer(const unsigned short* __restrict__ A1, int K1,
                     const unsigned short* __restrict__ B1,
                     const unsigned short* __restrict__ A2, int K2,
                     const unsigned short* __restrict__ B2,
                     const float* __restrict__ b_ih,
                     const float* __restrict__ b_hh,
                     float* __restrict__ c, float* __restrict__ h,
                     unsigned short* __restrict__ hb,
                     float* __restrict__ out, const int* __restrict__ x_len,
                     int t) {
  const int wave = threadIdx.x >> 5;
  const int lane = threadIdx.x & 31;
  const int nt   = blockIdx.x * 8 + wave;   // 0..127
  const int nl   = lane & 15;
  const bool lo  = lane < 16;

  v8f acc[4];
#pragma unroll
  for (int mt = 0; mt < 4; ++mt) acc[mt] = (v8f){0.f,0.f,0.f,0.f,0.f,0.f,0.f,0.f};

  auto mm = [&](const unsigned short* __restrict__ A, int K,
                const unsigned short* __restrict__ Bs) {
    const int KT = K >> 5;
    for (int kt = 0; kt < KT; ++kt) {
      // B fragment: 16 contiguous bf16 per lane (32B load)
      const v16bf b = *reinterpret_cast<const v16bf*>(
          Bs + (((size_t)nt * KT + kt) << 9) + (lane << 4));
      const int kbase = (kt << 5) + (lo ? 0 : 16);
#pragma unroll
      for (int mt = 0; mt < 4; ++mt) {
        // A fragment: row mt*16+nl, K runs [kbase..kbase+15]
        const unsigned short* ap = A + (size_t)(mt * 16 + nl) * K + kbase;
        v8bf a_lo = *reinterpret_cast<const v8bf*>(ap);
        v8bf a_hi = *reinterpret_cast<const v8bf*>(ap + 8);
        v16bf a = __builtin_shufflevector(a_lo, a_hi,
            0,1,2,3,4,5,6,7,8,9,10,11,12,13,14,15);
        acc[mt] = __builtin_amdgcn_wmma_f32_16x16x32_bf16(
            false, a, false, b, (short)0, acc[mt], false, false);
      }
    }
  };

  mm(A1, K1, B1);
  mm(A2, K2, B2);

  // ---- fused LSTM cell epilogue ----
  // Lane's own column: gate g = nl>>2, hidden j = nt*4 + (nl&3).
  const int j3    = lane & 3;
  const int half  = lane & 16;
  const int norig = (nl >> 2) * HH + nt * 4 + j3;
  const float bias = b_ih[norig] + b_hh[norig];
  const int jglob = nt * 4 + j3;
  const bool writer = (nl < 4);

#pragma unroll
  for (int mt = 0; mt < 4; ++mt) {
#pragma unroll
    for (int r = 0; r < 8; ++r) {
      float v = acc[mt][r] + bias;
      // gather the 4 gates for (row, jglob): source lanes half + g*4 + j3
      float gi = __shfl(v, half + 0  + j3, 32);
      float gf = __shfl(v, half + 4  + j3, 32);
      float gg = __shfl(v, half + 8  + j3, 32);
      float go = __shfl(v, half + 12 + j3, 32);
      int row = mt * 16 + r + (lo ? 0 : 8);
      int idx = row * HH + jglob;
      if (writer) {
        float cold = c[idx];
        float cn = __builtin_fmaf(hsigmoid(gf), cold,
                                  hsigmoid(gi) * htanh(gg));
        float hn = hsigmoid(go) * htanh(cn);
        c[idx] = cn;
        h[idx] = hn;
        hb[idx] = f32_to_bf16_rne(hn);
        if (out) out[(size_t)t * BH + idx] = (t < x_len[row]) ? hn : 0.f;
      }
    }
  }
}

// Final states/cells: d_out tail = stack([h0,h1]), stack([c0,c1])
__global__ void write_finals(const float* h0, const float* h1,
                             const float* c0, const float* c1,
                             float* __restrict__ out) {
  int idx = blockIdx.x * blockDim.x + threadIdx.x;  // 0..BH-1
  const size_t S = (size_t)TT * BH;                 // output elems
  out[S + idx]           = h0[idx];
  out[S + BH + idx]      = h1[idx];
  out[S + 2*BH + idx]    = c0[idx];
  out[S + 3*BH + idx]    = c1[idx];
}

extern "C" void kernel_launch(void* const* d_in, const int* in_sizes, int n_in,
                              void* d_out, int out_size, void* d_ws, size_t ws_size,
                              hipStream_t stream) {
  const float* x     = (const float*)d_in[0];
  const int*   x_len = (const int*)d_in[1];
  const float* W_ih0 = (const float*)d_in[2];
  const float* W_hh0 = (const float*)d_in[3];
  const float* b_ih0 = (const float*)d_in[4];
  const float* b_hh0 = (const float*)d_in[5];
  const float* W_ih1 = (const float*)d_in[6];
  const float* W_hh1 = (const float*)d_in[7];
  const float* b_ih1 = (const float*)d_in[8];
  const float* b_hh1 = (const float*)d_in[9];
  float* out = (float*)d_out;

  char* ws = (char*)d_ws;
  unsigned short* wih0 = (unsigned short*)(ws + OFF_WIH0);
  unsigned short* whh0 = (unsigned short*)(ws + OFF_WHH0);
  unsigned short* wih1 = (unsigned short*)(ws + OFF_WIH1);
  unsigned short* whh1 = (unsigned short*)(ws + OFF_WHH1);
  unsigned short* xb   = (unsigned short*)(ws + OFF_XB);
  float* h0 = (float*)(ws + OFF_H0);
  float* c0 = (float*)(ws + OFF_C0);
  float* h1 = (float*)(ws + OFF_H1);
  float* c1 = (float*)(ws + OFF_C1);
  unsigned short* h0b = (unsigned short*)(ws + OFF_H0B);
  unsigned short* h1b = (unsigned short*)(ws + OFF_H1B);

  // One-time conversions (re-run every call; deterministic, inputs untouched)
  convert_weight_swz<<<(G4*II + 255)/256, 256, 0, stream>>>(W_ih0, wih0, II);
  convert_weight_swz<<<(G4*HH + 255)/256, 256, 0, stream>>>(W_hh0, whh0, HH);
  convert_weight_swz<<<(G4*HH + 255)/256, 256, 0, stream>>>(W_ih1, wih1, HH);
  convert_weight_swz<<<(G4*HH + 255)/256, 256, 0, stream>>>(W_hh1, whh1, HH);
  convert_x_bf16<<<(TT*BB*II + 255)/256, 256, 0, stream>>>(x, xb, TT*BB*II);
  init_state<<<BH/256, 256, 0, stream>>>(h0, c0, h1, c1, h0b, h1b);

  for (int t = 0; t < TT; ++t) {
    const unsigned short* xt = xb + (size_t)t * BB * II;
    // layer 0: gates = x_t @ Wih0^T + h0 @ Whh0^T ; fused cell update
    lstm_step_layer<<<16, 256, 0, stream>>>(xt, II, wih0, h0b, HH, whh0,
                                            b_ih0, b_hh0, c0, h0, h0b,
                                            (float*)nullptr, x_len, t);
    // layer 1: gates = h0 @ Wih1^T + h1 @ Whh1^T ; fused + masked output
    lstm_step_layer<<<16, 256, 0, stream>>>(h0b, HH, wih1, h1b, HH, whh1,
                                            b_ih1, b_hh1, c1, h1, h1b,
                                            out, x_len, t);
  }

  write_finals<<<BH/256, 256, 0, stream>>>(h0, h1, c0, c1, out);
}
